// SelfAttention_18013092840154
// MI455X (gfx1250) — matrix operand order repaired and checked by reading, not compiled, not run on hardware
//
#include <hip/hip_runtime.h>
#include <hip/hip_bf16.h>

// ---------- types matching CDNA5 WMMA operand shapes ----------
typedef __attribute__((ext_vector_type(16))) __bf16       v16bf;
typedef __attribute__((ext_vector_type(8)))  float        v8f;
typedef unsigned int                                      u32;
typedef __attribute__((ext_vector_type(4)))  u32          u32x4;
typedef __attribute__((ext_vector_type(4)))  float        f32x4;

union ABu { v16bf v; u32x4 q[2]; };   // 8 VGPRs = one bf16 A/B operand

#define NPIX 4096   // H*W
#define CCH  512
#define DD   64

// native bf16 converts (gfx1250 has BF16 ALU; pairs fold to v_cvt_pk_bf16_f32)
__device__ __forceinline__ unsigned short bf16_1(float f) {
    return __builtin_bit_cast(unsigned short, (__bf16)f);
}
__device__ __forceinline__ u32 bf16_pk(float lo, float hi) {
    return (u32)bf16_1(lo) | ((u32)bf16_1(hi) << 16);
}

__device__ __forceinline__ v8f wmma_bf16(const ABu& a, const ABu& b, v8f c) {
    return __builtin_amdgcn_wmma_f32_16x16x32_bf16(
        false, a.v, false, b.v, (short)0, c, false, false);
}

// =====================================================================
// Kernel 1: fused Q/K/V projection  (relu(x@W + b)) -> bf16
//   Q,K stored [B][N][D] row-major; V stored transposed [B][D][N]
// =====================================================================
__global__ __launch_bounds__(256) void qkv_kernel(
    const float* __restrict__ x,
    const float* __restrict__ Wq, const float* __restrict__ bq,
    const float* __restrict__ Wk, const float* __restrict__ bk,
    const float* __restrict__ Wv, const float* __restrict__ bv,
    unsigned short* __restrict__ qo, unsigned short* __restrict__ ko,
    unsigned short* __restrict__ vto)
{
    const int blk = blockIdx.x;          // 256 blocks = B * (N/64)
    const int b   = blk >> 6;
    const int rb  = (blk & 63) * 64;     // pixel-row base within batch
    const float* xb = x + ((size_t)b * NPIX + rb) * CCH;

    __shared__ __align__(16) unsigned short Xs[64][40];      // 64 rows x 32 k (+pad)
    __shared__ __align__(16) unsigned short Ws[3][64][40];   // [mat][n][k] (+pad)

    const int tid  = threadIdx.x;
    const int wave = tid >> 5, lane = tid & 31;
    const int half = lane >> 4, l16 = lane & 15;
    const int rt   = wave >> 1;          // row tile 0..3 owned by this wave

    const float* Wmats[3]  = {Wq, Wk, Wv};
    const float* biases[3] = {bq, bk, bv};

    v8f acc[6];
    #pragma unroll
    for (int i = 0; i < 6; ++i) acc[i] = (v8f){};

    for (int kc = 0; kc < 16; ++kc) {    // 16 chunks of K=32 over C=512
        __syncthreads();
        // ---- stage x chunk [64][32] f32 -> bf16 LDS ----
        {
            int idx = tid * 8;
            int row = idx >> 5;
            int col = idx & 31;
            const f32x4* src = (const f32x4*)(xb + (size_t)row * CCH + kc * 32 + col);
            f32x4 f0 = src[0], f1 = src[1];
            u32x4 pk;
            pk.x = bf16_pk(f0.x, f0.y);
            pk.y = bf16_pk(f0.z, f0.w);
            pk.z = bf16_pk(f1.x, f1.y);
            pk.w = bf16_pk(f1.z, f1.w);
            *(u32x4*)&Xs[row][col] = pk;
        }
        // ---- stage W chunks transposed: Ws[mat][n][k] ----
        for (int e = tid; e < 6144; e += 256) {
            int mat = e >> 11;
            int r   = e & 2047;
            int k   = r >> 6, n = r & 63;
            Ws[mat][n][k] = bf16_1(Wmats[mat][(size_t)(kc * 32 + k) * DD + n]);
        }
        __syncthreads();

        // ---- issue ALL operand loads first, then the WMMA chain ----
        ABu a;
        {
            const unsigned short* base = &Xs[rt * 16 + l16][half * 8];
            a.q[0] = *(const u32x4*)base;
            a.q[1] = *(const u32x4*)(base + 16);
        }
        ABu bop[6];
        #pragma unroll
        for (int cc = 0; cc < 6; ++cc) {
            int combo = (wave & 1) * 6 + cc;   // 12 (mat,ct) combos split over wave parity
            int mat = combo >> 2, ct = combo & 3;
            const unsigned short* bb = &Ws[mat][ct * 16 + l16][half * 8];
            bop[cc].q[0] = *(const u32x4*)bb;
            bop[cc].q[1] = *(const u32x4*)(bb + 16);
        }
        #pragma unroll
        for (int cc = 0; cc < 6; ++cc)
            acc[cc] = wmma_bf16(a, bop[cc], acc[cc]);
    }

    // ---- epilogue: bias + relu + bf16 store ----
    #pragma unroll
    for (int cc = 0; cc < 6; ++cc) {
        int combo = (wave & 1) * 6 + cc;
        int mat = combo >> 2, ct = combo & 3;
        int col = ct * 16 + l16;
        float bias = biases[mat][col];
        #pragma unroll
        for (int r = 0; r < 8; ++r) {
            int row = rt * 16 + r + half * 8;
            float vv = acc[cc][r] + bias;
            vv = vv > 0.f ? vv : 0.f;
            unsigned short h = bf16_1(vv);
            size_t pix = (size_t)b * NPIX + rb + row;
            if (mat == 0)      qo[pix * DD + col] = h;
            else if (mat == 1) ko[pix * DD + col] = h;
            else               vto[((size_t)b * DD + col) * NPIX + rb + row] = h;
        }
    }
}

// =====================================================================
// Kernel 2: flash attention over 64-key blocks (online softmax, f32)
// =====================================================================
__global__ __launch_bounds__(256) void attn_kernel(
    const unsigned short* __restrict__ qg,
    const unsigned short* __restrict__ kg,
    const unsigned short* __restrict__ vt,
    unsigned short* __restrict__ ao)
{
    const int blk = blockIdx.x;          // 256 blocks = B * (N/64)
    const int b   = blk >> 6;
    const int qb  = (blk & 63) * 64;

    __shared__ __align__(16) unsigned short Qs[64][72];
    __shared__ __align__(16) float          Ss[64][66];
    __shared__ __align__(16) unsigned short Ps[64][72];
    __shared__ __align__(16) float          alphaS[64];
    __shared__ __align__(16) float          lS[64];

    const int tid  = threadIdx.x;
    const int wave = tid >> 5, lane = tid & 31;
    const int half = lane >> 4, l16 = lane & 15;
    const int rt   = wave >> 1;

    // ---- stage Q tile [64][64] bf16 ----
    {
        int row = tid >> 2, part = tid & 3;
        const u32x4* src = (const u32x4*)(qg + ((size_t)b * NPIX + qb + row) * DD + part * 16);
        u32x4 a0 = src[0], a1 = src[1];
        *(u32x4*)&Qs[row][part * 16]     = a0;
        *(u32x4*)&Qs[row][part * 16 + 8] = a1;
    }

    // A operands from Q are loop-invariant: hoist out of the key loop
    v8f o[2];
    o[0] = (v8f){}; o[1] = (v8f){};
    float mrow = -1e30f, lrow = 0.f;     // live in threads 0..63 only

    __syncthreads();
    ABu aq[2];
    #pragma unroll
    for (int ki = 0; ki < 2; ++ki) {
        const unsigned short* base = &Qs[rt * 16 + l16][ki * 32 + half * 8];
        aq[ki].q[0] = *(const u32x4*)base;
        aq[ki].q[1] = *(const u32x4*)(base + 16);
    }

    for (int kb = 0; kb < 64; ++kb) {
        // prefetch next key block's K and V^T tiles into cache (global_prefetch_b8)
        if (kb + 1 < 64) {
            const unsigned short* nk = kg + ((size_t)b * NPIX + (kb + 1) * 64) * DD;
            const unsigned short* nv = vt + (size_t)b * DD * NPIX + (kb + 1) * 64;
            __builtin_prefetch(nk + (size_t)tid * 32, 0, 0);            // 128 rows x 128B
            __builtin_prefetch(nv + (size_t)(tid & 63) * NPIX, 0, 0);   // 64 d-rows x 128B
        }
        // ---- S = Q * K^T : load all 4 B operands, then 4 WMMAs ----
        ABu bk_[2][2];
        #pragma unroll
        for (int c = 0; c < 2; ++c) {
            int ct = (wave & 1) * 2 + c;
            #pragma unroll
            for (int ki = 0; ki < 2; ++ki) {
                const unsigned short* kp =
                    kg + ((size_t)b * NPIX + kb * 64 + ct * 16 + l16) * DD + ki * 32 + half * 8;
                bk_[c][ki].q[0] = *(const u32x4*)kp;
                bk_[c][ki].q[1] = *(const u32x4*)(kp + 16);
            }
        }
        v8f s[2];
        #pragma unroll
        for (int c = 0; c < 2; ++c) {
            s[c] = (v8f){};
            #pragma unroll
            for (int ki = 0; ki < 2; ++ki)
                s[c] = wmma_bf16(aq[ki], bk_[c][ki], s[c]);
        }
        #pragma unroll
        for (int c = 0; c < 2; ++c) {
            int ct = (wave & 1) * 2 + c;
            #pragma unroll
            for (int r = 0; r < 8; ++r)
                Ss[rt * 16 + r + half * 8][ct * 16 + l16] = s[c][r];
        }
        __syncthreads();
        // ---- online softmax, one thread per query row ----
        if (tid < 64) {
            float mb = -1e30f;
            #pragma unroll 8
            for (int j = 0; j < 64; ++j) mb = fmaxf(mb, Ss[tid][j]);
            float mnew  = fmaxf(mrow, mb);
            float alpha = __expf(mrow - mnew);
            float rs = 0.f;
            #pragma unroll 4
            for (int j = 0; j < 64; j += 2) {
                float p0 = __expf(Ss[tid][j]     - mnew);
                float p1 = __expf(Ss[tid][j + 1] - mnew);
                rs += p0 + p1;
                *(u32*)&Ps[tid][j] = bf16_pk(p0, p1);
            }
            lrow = lrow * alpha + rs;
            mrow = mnew;
            alphaS[tid] = alpha;
        }
        __syncthreads();
        // ---- O = O*alpha + P * V : load all operands, then WMMA chain ----
        ABu ap[2];
        #pragma unroll
        for (int ki = 0; ki < 2; ++ki) {
            const unsigned short* base = &Ps[rt * 16 + l16][ki * 32 + half * 8];
            ap[ki].q[0] = *(const u32x4*)base;
            ap[ki].q[1] = *(const u32x4*)(base + 16);
        }
        ABu bv_[2][2];
        #pragma unroll
        for (int c = 0; c < 2; ++c) {
            int ct = (wave & 1) * 2 + c;
            #pragma unroll
            for (int ki = 0; ki < 2; ++ki) {
                const unsigned short* vp =
                    vt + ((size_t)b * DD + ct * 16 + l16) * NPIX + kb * 64 + ki * 32 + half * 8;
                bv_[c][ki].q[0] = *(const u32x4*)vp;
                bv_[c][ki].q[1] = *(const u32x4*)(vp + 16);
            }
        }
        float av0 = alphaS[rt * 16 + half * 8];      // rows r+half*8, r=0..7
        #pragma unroll
        for (int c = 0; c < 2; ++c) {
            #pragma unroll
            for (int r = 0; r < 8; ++r)
                o[c][r] = o[c][r] * alphaS[rt * 16 + r + half * 8];
            #pragma unroll
            for (int ki = 0; ki < 2; ++ki)
                o[c] = wmma_bf16(ap[ki], bv_[c][ki], o[c]);
        }
        (void)av0;
        __syncthreads();   // protect Ss/Ps reuse next iteration
    }

    if (tid < 64) lS[tid] = lrow;
    __syncthreads();
    #pragma unroll
    for (int c = 0; c < 2; ++c) {
        int ct = (wave & 1) * 2 + c;
        #pragma unroll
        for (int r = 0; r < 8; ++r) {
            int row = rt * 16 + r + half * 8;
            float vv = o[c][r] / lS[row];
            ao[((size_t)b * NPIX + qb + row) * DD + ct * 16 + l16] = bf16_1(vv);
        }
    }
}

// =====================================================================
// Kernel 3: output projection  relu(attn @ Wo + bo) -> f32
// =====================================================================
__global__ __launch_bounds__(256) void proj_kernel(
    const unsigned short* __restrict__ ao,
    const float* __restrict__ Wo, const float* __restrict__ bo,
    float* __restrict__ out)
{
    const int blk = blockIdx.x;          // 256 blocks of 64 rows
    const size_t rbase = (size_t)blk * 64;

    __shared__ __align__(16) unsigned short As[64][72];
    __shared__ __align__(16) unsigned short Wos[128][72];  // [c][d] chunk

    const int tid  = threadIdx.x;
    const int wave = tid >> 5, lane = tid & 31;
    const int half = lane >> 4, l16 = lane & 15;
    const int rt   = wave >> 1;

    // stage A rows [64][64] bf16
    {
        int row = tid >> 2, part = tid & 3;
        const u32x4* src = (const u32x4*)(ao + (rbase + row) * DD + part * 16);
        u32x4 a0 = src[0], a1 = src[1];
        *(u32x4*)&As[row][part * 16]     = a0;
        *(u32x4*)&As[row][part * 16 + 8] = a1;
    }

    for (int cb = 0; cb < 4; ++cb) {     // 4 column chunks of 128 over C=512
        __syncthreads();
        // stage Wo chunk transposed: Wos[c][d]
        for (int e = tid; e < 8192; e += 256) {
            int d = e >> 7, c = e & 127;
            Wos[c][d] = bf16_1(Wo[(size_t)d * CCH + cb * 128 + c]);
        }
        __syncthreads();

        ABu a[2];
        #pragma unroll
        for (int ki = 0; ki < 2; ++ki) {
            const unsigned short* base = &As[rt * 16 + l16][ki * 32 + half * 8];
            a[ki].q[0] = *(const u32x4*)base;
            a[ki].q[1] = *(const u32x4*)(base + 16);
        }
        #pragma unroll
        for (int c = 0; c < 4; ++c) {
            int ct = (wave & 1) * 4 + c;
            // load both K-iteration B operands, then both WMMAs
            ABu bw[2];
            #pragma unroll
            for (int ki = 0; ki < 2; ++ki) {
                const unsigned short* wb = &Wos[ct * 16 + l16][ki * 32 + half * 8];
                bw[ki].q[0] = *(const u32x4*)wb;
                bw[ki].q[1] = *(const u32x4*)(wb + 16);
            }
            v8f acc = (v8f){};
            #pragma unroll
            for (int ki = 0; ki < 2; ++ki)
                acc = wmma_bf16(a[ki], bw[ki], acc);
            int col = cb * 128 + ct * 16 + l16;
            float bias = bo[col];
            #pragma unroll
            for (int r = 0; r < 8; ++r) {
                int row = rt * 16 + r + half * 8;
                float vv = acc[r] + bias;
                out[(rbase + row) * CCH + col] = vv > 0.f ? vv : 0.f;
            }
        }
    }
}

// =====================================================================
extern "C" void kernel_launch(void* const* d_in, const int* in_sizes, int n_in,
                              void* d_out, int out_size, void* d_ws, size_t ws_size,
                              hipStream_t stream) {
    const float* x  = (const float*)d_in[0];
    const float* Wq = (const float*)d_in[1];
    const float* bq = (const float*)d_in[2];
    const float* Wk = (const float*)d_in[3];
    const float* bk = (const float*)d_in[4];
    const float* Wv = (const float*)d_in[5];
    const float* bv = (const float*)d_in[6];
    const float* Wo = (const float*)d_in[7];
    const float* bo = (const float*)d_in[8];
    float* out = (float*)d_out;

    const size_t nqkv = (size_t)4 * NPIX * DD;   // 1M elems each
    unsigned short* qw  = (unsigned short*)d_ws;
    unsigned short* kw  = qw  + nqkv;
    unsigned short* vtw = kw  + nqkv;
    unsigned short* aow = vtw + nqkv;

    qkv_kernel<<<256, 256, 0, stream>>>(x, Wq, bq, Wk, bk, Wv, bv, qw, kw, vtw);
    attn_kernel<<<256, 256, 0, stream>>>(qw, kw, vtw, aow);
    proj_kernel<<<256, 256, 0, stream>>>(aow, Wo, bo, out);
}
